// Quant_vgg9_55078660604331
// MI455X (gfx1250) — 8-bit matrix operand order repaired and checked by reading, compiled, not run on hardware
//
#include <hip/hip_runtime.h>
#include <cstdint>
#include <cstddef>

typedef __attribute__((ext_vector_type(8))) int v8i;

#define BN_EPS 1e-5f

__global__ __launch_bounds__(256)
void conv0_kernel(const float* __restrict__ x, const float* __restrict__ w,
                  float* __restrict__ out, int N, int H, int W, int Cin, int Cout)
{
    int idx = blockIdx.x * blockDim.x + threadIdx.x;
    int total = N * H * W * Cout;
    if (idx >= total) return;
    int co  = idx % Cout;
    int pix = idx / Cout;
    int n = pix / (H * W);
    int r = pix - n * H * W;
    int y = r / W;
    int xx0 = r - y * W;
    float acc = 0.f;
    for (int ci = 0; ci < Cin; ++ci) {
        for (int ky = 0; ky < 3; ++ky) {
            int yy = y + ky - 1;
            if (yy < 0 || yy >= H) continue;
            for (int kx = 0; kx < 3; ++kx) {
                int xc = xx0 + kx - 1;
                if (xc < 0 || xc >= W) continue;
                acc += x[(((size_t)n * Cin + ci) * H + yy) * W + xc] *
                       w[((co * Cin + ci) * 3 + ky) * 3 + kx];
            }
        }
    }
    out[(size_t)pix * Cout + co] = acc;
}

template <bool IS_CONV>
__global__ __launch_bounds__(256)
void gemm_iu8_kernel(const unsigned char* __restrict__ A,
                     const signed char* __restrict__ B,
                     float* __restrict__ O,
                     int M, int K, int Nout,
                     int H, int W, int Cin, float scale)
{
    __shared__ __align__(16) unsigned char At[2][32 * 64];
    __shared__ __align__(16) signed char   Bt[2][128 * 64];

    const int tid   = threadIdx.x;
    const int wave  = tid >> 5;
    const int lane  = tid & 31;
    const int waveM = wave & 1;
    const int waveN = wave >> 1;
    const int mbase = blockIdx.x * 32;
    const int nbase = blockIdx.y * 128;

    const int pA  = tid >> 3;
    const int byA = (tid & 7) * 8;
    const int gp  = mbase + pA;
    const int colB = tid >> 1;
    const int byB  = (tid & 1) * 32;

    const int arow  = waveM * 16 + (lane & 15);
    const int aoff  = arow * 64 + ((lane & 16) ? 16 : 0);
    const int bcol0 = waveN * 32 + (lane & 15);
    const int boff0 = bcol0 * 64 + ((lane & 16) ? 16 : 0);
    const int boff1 = boff0 + 16 * 64;

    int n = 0, y = 0, x = 0;
    if (IS_CONV) {
        n = gp / (H * W);
        int r = gp - n * H * W;
        y = r / W;
        x = r - y * W;
    }
    const int cin64 = IS_CONV ? (Cin >> 6) : 0;
    int ky = 0, kx = 0, cc = 0;
    const unsigned char* aptr;
    bool avalid;
    if (IS_CONV) {
        int yy = y - 1, xc = x - 1;
        avalid = (yy >= 0) && (yy < H) && (xc >= 0) && (xc < W);
        aptr = A + (((size_t)n * H + yy) * W + xc) * Cin + byA;
    } else {
        avalid = true;
        aptr = A + (size_t)gp * K + byA;
    }
    const signed char* bptr = B + (size_t)(nbase + colB) * K + byB;

    auto advanceA = [&]() {
        if (IS_CONV) {
            aptr += 64;
            if (++cc == cin64) {
                cc = 0;
                if (++kx == 3) { kx = 0; ++ky; }
                int yy = y + ky - 1, xc = x + kx - 1;
                avalid = (yy >= 0) && (yy < H) && (xc >= 0) && (xc < W);
                aptr = A + (((size_t)n * H + yy) * W + xc) * Cin + byA;
            }
        } else {
            aptr += 64;
        }
    };

    v8i acc0 = {0, 0, 0, 0, 0, 0, 0, 0};
    v8i acc1 = {0, 0, 0, 0, 0, 0, 0, 0};
    const int nChunks = K / 64;

    {
        int2 a0 = make_int2(0, 0);
        if (avalid) a0 = *reinterpret_cast<const int2*>(aptr);
        int4 b0 = *reinterpret_cast<const int4*>(bptr);
        int4 b1 = *reinterpret_cast<const int4*>(bptr + 16);
        *reinterpret_cast<int2*>(&At[0][pA * 64 + byA])        = a0;
        *reinterpret_cast<int4*>(&Bt[0][colB * 64 + byB])      = b0;
        *reinterpret_cast<int4*>(&Bt[0][colB * 64 + byB + 16]) = b1;
        advanceA();
        bptr += 64;
    }
    __syncthreads();

    for (int kc = 0; kc < nChunks; ++kc) {
        const int  cur     = kc & 1;
        const bool hasNext = (kc + 1 < nChunks);

        int2 aN  = make_int2(0, 0);
        int4 bN0 = make_int4(0, 0, 0, 0);
        int4 bN1 = make_int4(0, 0, 0, 0);
        if (hasNext) {
            if (!IS_CONV || avalid) aN = *reinterpret_cast<const int2*>(aptr);
            bN0 = *reinterpret_cast<const int4*>(bptr);
            bN1 = *reinterpret_cast<const int4*>(bptr + 16);
            __builtin_prefetch(bptr + 64, 0, 3);
        }

        v8i a, b0, b1;
        {
            const unsigned char* Ab = &At[cur][aoff];
            int4 q0 = *reinterpret_cast<const int4*>(Ab);
            int4 q1 = *reinterpret_cast<const int4*>(Ab + 32);
            a[0] = q0.x; a[1] = q0.y; a[2] = q0.z; a[3] = q0.w;
            a[4] = q1.x; a[5] = q1.y; a[6] = q1.z; a[7] = q1.w;
        }
        {
            const signed char* Bb = &Bt[cur][boff0];
            int4 r0 = *reinterpret_cast<const int4*>(Bb);
            int4 r1 = *reinterpret_cast<const int4*>(Bb + 32);
            b0[0] = r0.x; b0[1] = r0.y; b0[2] = r0.z; b0[3] = r0.w;
            b0[4] = r1.x; b0[5] = r1.y; b0[6] = r1.z; b0[7] = r1.w;
        }
        {
            const signed char* Bb = &Bt[cur][boff1];
            int4 r0 = *reinterpret_cast<const int4*>(Bb);
            int4 r1 = *reinterpret_cast<const int4*>(Bb + 32);
            b1[0] = r0.x; b1[1] = r0.y; b1[2] = r0.z; b1[3] = r0.w;
            b1[4] = r1.x; b1[5] = r1.y; b1[6] = r1.z; b1[7] = r1.w;
        }
        acc0 = __builtin_amdgcn_wmma_i32_16x16x64_iu8(false, a, true, b0, acc0,
                                                      false, false);
        acc1 = __builtin_amdgcn_wmma_i32_16x16x64_iu8(false, a, true, b1, acc1,
                                                      false, false);

        if (hasNext) {
            *reinterpret_cast<int2*>(&At[cur ^ 1][pA * 64 + byA])        = aN;
            *reinterpret_cast<int4*>(&Bt[cur ^ 1][colB * 64 + byB])      = bN0;
            *reinterpret_cast<int4*>(&Bt[cur ^ 1][colB * 64 + byB + 16]) = bN1;
            advanceA();
            bptr += 64;
        }
        __syncthreads();
    }

    const int ocol0    = nbase + waveN * 32 + (lane & 15);
    const int orowbase = mbase + waveM * 16 + ((lane & 16) ? 8 : 0);
#pragma unroll
    for (int r = 0; r < 8; ++r) {
        O[(size_t)(orowbase + r) * Nout + ocol0]      = (float)acc0[r] * scale;
        O[(size_t)(orowbase + r) * Nout + ocol0 + 16] = (float)acc1[r] * scale;
    }
}

__global__ __launch_bounds__(256)
void stats_kernel(const float* __restrict__ v, float* __restrict__ sum,
                  float* __restrict__ sumsq, size_t total, int C)
{
    extern __shared__ float sh[];
    float* s0 = sh;
    float* s1 = sh + C;
    for (int c = threadIdx.x; c < 2 * C; c += blockDim.x) sh[c] = 0.f;
    __syncthreads();
    size_t stride = (size_t)gridDim.x * blockDim.x;
    for (size_t i = (size_t)blockIdx.x * blockDim.x + threadIdx.x; i < total; i += stride) {
        float val = v[i];
        int c = (int)(i & (size_t)(C - 1));
        atomicAdd(&s0[c], val);
        atomicAdd(&s1[c], val * val);
    }
    __syncthreads();
    for (int c = threadIdx.x; c < C; c += blockDim.x) {
        atomicAdd(&sum[c], s0[c]);
        atomicAdd(&sumsq[c], s1[c]);
    }
}

__global__ void bnfin_kernel(const float* __restrict__ sum, const float* __restrict__ sumsq,
                             const float* __restrict__ gb, float* __restrict__ scsh,
                             int C, float invP)
{
    int c = blockIdx.x * blockDim.x + threadIdx.x;
    if (c >= C) return;
    float m   = sum[c] * invP;
    float var = sumsq[c] * invP - m * m;
    float sc  = gb[c] * rsqrtf(var + BN_EPS);
    scsh[c]     = sc;
    scsh[C + c] = gb[C + c] - m * sc;
}

__global__ __launch_bounds__(256)
void bnquant_kernel(const float* __restrict__ v, const float* __restrict__ scsh,
                    unsigned char* __restrict__ out, size_t total, int C)
{
    size_t i = (size_t)blockIdx.x * blockDim.x + threadIdx.x;
    if (i >= total) return;
    int c = (int)(i & (size_t)(C - 1));
    float val = v[i] * scsh[c] + scsh[C + c];
    val = fminf(fmaxf(val, 0.f), 1.f);
    out[i] = (unsigned char)(int)rintf(val * 15.f);
}

__global__ __launch_bounds__(256)
void bnrelu_kernel(const float* __restrict__ v, const float* __restrict__ scsh,
                   float* __restrict__ out, size_t total, int C)
{
    size_t i = (size_t)blockIdx.x * blockDim.x + threadIdx.x;
    if (i >= total) return;
    int c = (int)(i & (size_t)(C - 1));
    out[i] = fmaxf(v[i] * scsh[c] + scsh[C + c], 0.f);
}

__global__ __launch_bounds__(256)
void maxpool_kernel(const float* __restrict__ in, float* __restrict__ out,
                    int N, int H, int W, int C)
{
    int Ho = H >> 1, Wo = W >> 1;
    size_t total = (size_t)N * Ho * Wo * C;
    size_t idx = (size_t)blockIdx.x * blockDim.x + threadIdx.x;
    if (idx >= total) return;
    int c   = (int)(idx & (size_t)(C - 1));
    size_t pix = idx / C;
    int n = (int)(pix / (Ho * Wo));
    int r = (int)(pix - (size_t)n * Ho * Wo);
    int y = r / Wo, x = r - y * Wo;
    size_t base = (((size_t)n * H + 2 * y) * W + 2 * x) * C + c;
    float a0 = in[base],                 a1 = in[base + C];
    float a2 = in[base + (size_t)W * C], a3 = in[base + (size_t)W * C + C];
    out[idx] = fmaxf(fmaxf(a0, a1), fmaxf(a2, a3));
}

__global__ __launch_bounds__(256)
void qconvw_kernel(const float* __restrict__ w, signed char* __restrict__ out,
                   int Co, int Ci)
{
    int idx = blockIdx.x * blockDim.x + threadIdx.x;
    int total = Co * Ci * 9;
    if (idx >= total) return;
    int co  = idx / (Ci * 9);
    int r   = idx - co * Ci * 9;
    int ci  = r / 9;
    int tap = r - ci * 9;
    float q = rintf(fminf(fmaxf(w[idx], -1.f), 1.f) * 7.f);
    out[(size_t)co * 9 * Ci + tap * Ci + ci] = (signed char)(int)q;
}

__global__ __launch_bounds__(256)
void qfcw_kernel(const float* __restrict__ w, signed char* __restrict__ out,
                 int Co, int K, int permute)
{
    int idx = blockIdx.x * blockDim.x + threadIdx.x;
    if (idx >= Co * K) return;
    int o = idx / K;
    int k = idx - o * K;
    int incol = k;
    if (permute) {
        int s = k >> 9;
        int c = k & 511;
        incol = c * 16 + s;
    }
    float q = rintf(fminf(fmaxf(w[(size_t)o * K + incol], -1.f), 1.f) * 7.f);
    out[idx] = (signed char)(int)q;
}

__global__ __launch_bounds__(256)
void fc2_kernel(const float* __restrict__ h, const float* __restrict__ w,
                const float* __restrict__ bias, float* __restrict__ out,
                int M, int K, int Nout)
{
    int idx = blockIdx.x * blockDim.x + threadIdx.x;
    if (idx >= M * Nout) return;
    int col = idx % Nout;
    int row = idx / Nout;
    float acc = bias[col];
    for (int k = 0; k < K; ++k)
        acc += h[(size_t)row * K + k] * w[(size_t)col * K + k];
    out[idx] = acc;
}

__global__ __launch_bounds__(256)
void bn8_kernel(const float* __restrict__ v, const float* __restrict__ gb,
                float* __restrict__ out, int M, int C)
{
    __shared__ float s0[256];
    __shared__ float s1[256];
    int c = blockIdx.x;
    int r = threadIdx.x;
    float val = v[(size_t)r * C + c];
    s0[r] = val;
    s1[r] = val * val;
    __syncthreads();
    for (int o = 128; o > 0; o >>= 1) {
        if (r < o) { s0[r] += s0[r + o]; s1[r] += s1[r + o]; }
        __syncthreads();
    }
    float m   = s0[0] / M;
    float var = s1[0] / M - m * m;
    float sc  = gb[c] * rsqrtf(var + BN_EPS);
    float shf = gb[C + c] - m * sc;
    out[(size_t)r * C + c] = val * sc + shf;
}

extern "C" void kernel_launch(void* const* d_in, const int* in_sizes, int n_in,
                              void* d_out, int out_size, void* d_ws, size_t ws_size,
                              hipStream_t stream)
{
    (void)in_sizes; (void)n_in; (void)out_size; (void)ws_size;

    const float* x   = (const float*)d_in[0];
    const float* cw[6] = {(const float*)d_in[1], (const float*)d_in[2],
                          (const float*)d_in[3], (const float*)d_in[4],
                          (const float*)d_in[5], (const float*)d_in[6]};
    const float* fw0 = (const float*)d_in[7];
    const float* fw1 = (const float*)d_in[8];
    const float* fw2 = (const float*)d_in[9];
    const float* fb2 = (const float*)d_in[10];
    const float* bn[9];
    for (int i = 0; i < 9; ++i) bn[i] = (const float*)d_in[11 + i];
    float* out = (float*)d_out;

    char* ws = (char*)d_ws;
    size_t off = 0;
    auto alloc = [&](size_t bytes) {
        size_t r = off;
        off += (bytes + 255) & ~(size_t)255;
        return r;
    };
    float*         F0   = (float*)(ws + alloc(134217728));
    float*         F1   = (float*)(ws + alloc(33554432));
    unsigned char* ACTA = (unsigned char*)(ws + alloc(33554432));
    unsigned char* ACTB = (unsigned char*)(ws + alloc(33554432));
    signed char*   WC1  = (signed char*)(ws + alloc(128 * 1152));
    signed char*   WC2  = (signed char*)(ws + alloc(256 * 1152));
    signed char*   WC3  = (signed char*)(ws + alloc(256 * 2304));
    signed char*   WC4  = (signed char*)(ws + alloc(512 * 2304));
    signed char*   WC5  = (signed char*)(ws + alloc(512 * 4608));
    signed char*   WF0  = (signed char*)(ws + alloc(8388608));
    signed char*   WF1  = (signed char*)(ws + alloc(1048576));
    float*         SUM  = (float*)(ws + alloc(4096));
    float*         SUMQ = (float*)(ws + alloc(4096));
    float*         SCSH = (float*)(ws + alloc(8192));

    const float convScale = 1.0f / 105.0f;
    const int   N = 256;

    auto bn_pass = [&](const float* buf, size_t P, int C, const float* gb,
                       unsigned char* qout, float* fout) {
        hipMemsetAsync(SUM, 0, 8192, stream);
        size_t total = P * (size_t)C;
        stats_kernel<<<2048, 256, 2 * C * sizeof(float), stream>>>(buf, SUM, SUMQ, total, C);
        bnfin_kernel<<<(C + 255) / 256, 256, 0, stream>>>(SUM, SUMQ, gb, SCSH, C, 1.0f / (float)P);
        int blocks = (int)((total + 255) / 256);
        if (qout)
            bnquant_kernel<<<blocks, 256, 0, stream>>>(buf, SCSH, qout, total, C);
        else
            bnrelu_kernel<<<blocks, 256, 0, stream>>>(buf, SCSH, fout, total, C);
    };

    {
        const int ci[5] = {128, 128, 256, 256, 512};
        const int co[5] = {128, 256, 256, 512, 512};
        signed char* wq[5] = {WC1, WC2, WC3, WC4, WC5};
        for (int i = 0; i < 5; ++i) {
            int tot = co[i] * ci[i] * 9;
            qconvw_kernel<<<(tot + 255) / 256, 256, 0, stream>>>(cw[i + 1], wq[i], co[i], ci[i]);
        }
        qfcw_kernel<<<(1024 * 8192) / 256, 256, 0, stream>>>(fw0, WF0, 1024, 8192, 1);
        qfcw_kernel<<<(1024 * 1024) / 256, 256, 0, stream>>>(fw1, WF1, 1024, 1024, 0);
    }

    conv0_kernel<<<(N * 32 * 32 * 128 + 255) / 256, 256, 0, stream>>>(x, cw[0], F0, N, 32, 32, 3, 128);
    bn_pass(F0, (size_t)N * 1024, 128, bn[0], ACTA, nullptr);

    gemm_iu8_kernel<true><<<dim3(N * 1024 / 32, 128 / 128), 256, 0, stream>>>(
        ACTA, WC1, F0, N * 1024, 1152, 128, 32, 32, 128, convScale);
    maxpool_kernel<<<(int)(((size_t)N * 256 * 128 + 255) / 256), 256, 0, stream>>>(F0, F1, N, 32, 32, 128);
    bn_pass(F1, (size_t)N * 256, 128, bn[1], ACTB, nullptr);

    gemm_iu8_kernel<true><<<dim3(N * 256 / 32, 256 / 128), 256, 0, stream>>>(
        ACTB, WC2, F0, N * 256, 1152, 256, 16, 16, 128, convScale);
    bn_pass(F0, (size_t)N * 256, 256, bn[2], ACTA, nullptr);

    gemm_iu8_kernel<true><<<dim3(N * 256 / 32, 256 / 128), 256, 0, stream>>>(
        ACTA, WC3, F0, N * 256, 2304, 256, 16, 16, 256, convScale);
    maxpool_kernel<<<(int)(((size_t)N * 64 * 256 + 255) / 256), 256, 0, stream>>>(F0, F1, N, 16, 16, 256);
    bn_pass(F1, (size_t)N * 64, 256, bn[3], ACTB, nullptr);

    gemm_iu8_kernel<true><<<dim3(N * 64 / 32, 512 / 128), 256, 0, stream>>>(
        ACTB, WC4, F0, N * 64, 2304, 512, 8, 8, 256, convScale);
    bn_pass(F0, (size_t)N * 64, 512, bn[4], ACTA, nullptr);

    gemm_iu8_kernel<true><<<dim3(N * 64 / 32, 512 / 128), 256, 0, stream>>>(
        ACTA, WC5, F0, N * 64, 4608, 512, 8, 8, 512, convScale);
    maxpool_kernel<<<(int)(((size_t)N * 16 * 512 + 255) / 256), 256, 0, stream>>>(F0, F1, N, 8, 8, 512);
    bn_pass(F1, (size_t)N * 16, 512, bn[5], ACTB, nullptr);

    gemm_iu8_kernel<false><<<dim3(N / 32, 1024 / 128), 256, 0, stream>>>(
        ACTB, WF0, F0, N, 8192, 1024, 0, 0, 0, convScale);
    bn_pass(F0, (size_t)N, 1024, bn[6], ACTA, nullptr);

    gemm_iu8_kernel<false><<<dim3(N / 32, 1024 / 128), 256, 0, stream>>>(
        ACTA, WF1, F0, N, 1024, 1024, 0, 0, 0, convScale);
    bn_pass(F0, (size_t)N, 1024, bn[7], nullptr, F1);

    fc2_kernel<<<(N * 10 + 255) / 256, 256, 0, stream>>>(F1, fw2, fb2, F0, N, 1024, 10);
    bn8_kernel<<<10, 256, 0, stream>>>(F0, bn[8], out, N, 10);
}
